// SoftDecisionTree_35716948033588
// MI455X (gfx1250) — matrix operand order repaired and checked by reading, not compiled, not run on hardware
//
#include <hip/hip_runtime.h>
#include <hip/hip_bf16.h>
#include <math.h>

typedef __attribute__((ext_vector_type(16))) __bf16 v16bf;
typedef __attribute__((ext_vector_type(8)))  __bf16 v8bf;
typedef __attribute__((ext_vector_type(8)))  float  v8f;
typedef __attribute__((ext_vector_type(4)))  float  v4f;

#define B_TOTAL 131072
#define D_DIM   64
#define C_DIM   16
#define N_INNER 1023
#define N_LEAF  1024

#define TB          32     // rows per block
#define P_STRIDE    1028   // f32 stride, p tile   (1028 mod 64 = 4 -> conflict-free lanes)
#define PATH_STRIDE 1032   // bf16 stride, path    (516 words, 4-bank rotation per row)
#define QT_STRIDE   1032   // bf16 stride, Q^T
#define X_STRIDE    72     // bf16 stride, X tile

// workspace layout (bytes)
#define WS_QT_OFF    0                      // bf16 Q^T [16][1024]            = 32768 B
#define WS_BIAS_OFF  32768                  // f32 beta*b [1024]              =  4096 B
#define WS_WB_OFF    36864                  // bf16 beta*W [1024][64]         = 131072 B

__device__ __forceinline__ unsigned short f2bfbits(float f) {
    union { float f; unsigned u; } x; x.f = f;
    unsigned r = x.u + 0x7FFFu + ((x.u >> 16) & 1u);   // round-to-nearest-even
    return (unsigned short)(r >> 16);
}
__device__ __forceinline__ __bf16 f2bf(float f) {
    return __builtin_bit_cast(__bf16, f2bfbits(f));
}

// ---------------------------------------------------------------------------
// Prep: Q^T = bf16(softmax(leaf_params))^T ; biasp = beta*b ; Wb = bf16(beta*W)
// ---------------------------------------------------------------------------
__global__ void sdt_prep_kernel(const float* __restrict__ W,
                                const float* __restrict__ b,
                                const float* __restrict__ beta,
                                const float* __restrict__ leaf,
                                unsigned short* __restrict__ qt,   // [16][1024] bf16
                                float* __restrict__ biasp,
                                __bf16* __restrict__ wb) {
    int tid = blockIdx.x * blockDim.x + threadIdx.x;
    int nth = gridDim.x * blockDim.x;

    if (tid < N_LEAF) {
        float v[C_DIM];
        float m = -INFINITY;
        #pragma unroll
        for (int i = 0; i < C_DIM; ++i) { v[i] = leaf[tid * C_DIM + i]; m = fmaxf(m, v[i]); }
        float s = 0.f;
        #pragma unroll
        for (int i = 0; i < C_DIM; ++i) { v[i] = __expf(v[i] - m); s += v[i]; }
        float inv = 1.f / s;
        #pragma unroll
        for (int i = 0; i < C_DIM; ++i) qt[i * N_LEAF + tid] = f2bfbits(v[i] * inv);

        biasp[tid] = (tid < N_INNER) ? beta[tid] * b[tid] : 0.f;
    }

    for (int i = tid; i < N_LEAF * D_DIM; i += nth) {
        int n = i >> 6;
        int k = i & 63;
        float val = (n < N_INNER) ? W[n * D_DIM + k] * beta[n] : 0.f;
        wb[i] = f2bf(val);
    }
}

// ---------------------------------------------------------------------------
// Fused main kernel, 32 rows / block, 256 threads (8 waves):
//   Phase 1: p = sigmoid(Xb @ Wb^T + bias)  via v_wmma_f32_16x16x32_bf16 -> LDS
//   Phase 2: level-DP path expansion (depth-5 subtrees in registers)     -> LDS bf16
//   Phase 3: out = path @ Q                 via v_wmma_f32_16x16x32_bf16
// ---------------------------------------------------------------------------
__global__ void __launch_bounds__(256)
sdt_main_kernel(const float* __restrict__ X,
                const __bf16* __restrict__ qtg,   // bf16 Q^T [16][1024]
                const float* __restrict__ biasp,
                const __bf16* __restrict__ wb,
                float* __restrict__ out) {
    extern __shared__ char smem[];
    float*  lds_p    = (float*)smem;                        // [32][1028] f32   131584 B
    __bf16* lds_path = (__bf16*)(smem + 131584);            // [32][1032] bf16   66048 B
    __bf16* lds_qt   = (__bf16*)(smem + 197632);            // [16][1032] bf16   33024 B
    __bf16* lds_x    = (__bf16*)(smem + 230656);            // [32][72]   bf16    4608 B
    float*  lds_o    = (float*)(smem + 235264);             // [32][16]   f32     2048 B

    const int t    = threadIdx.x;
    const int row0 = blockIdx.x * TB;
    const int w    = t >> 5;
    const int lane = t & 31;
    const int mcol = lane & 15;
    const bool hi  = lane >= 16;

    // zero mixture accumulators
    for (int i = t; i < TB * C_DIM; i += 256) lds_o[i] = 0.f;

    // stage X tile (f32 global -> bf16 LDS), 8 elems/thread
    {
        int r  = t >> 3;
        int k0 = (t & 7) * 8;
        const float* src = X + (size_t)(row0 + r) * D_DIM + k0;
        v4f a = *(const v4f*)src;
        v4f c = *(const v4f*)(src + 4);
        __bf16* dst = lds_x + r * X_STRIDE + k0;
        dst[0] = f2bf(a.x); dst[1] = f2bf(a.y); dst[2] = f2bf(a.z); dst[3] = f2bf(a.w);
        dst[4] = f2bf(c.x); dst[5] = f2bf(c.y); dst[6] = f2bf(c.z); dst[7] = f2bf(c.w);
    }

    // stage Q^T (bf16), 4 elems per chunk
    {
        const uint2* qsrc = (const uint2*)qtg;
        for (int ci = t; ci < (C_DIM * N_LEAF) / 4; ci += 256) {
            int c  = ci >> 8;           // 1024/4 = 256 chunks per class
            int l0 = (ci & 255) * 4;
            *(uint2*)(lds_qt + c * QT_STRIDE + l0) = qsrc[ci];
        }
    }
    __syncthreads();

    // ------------------ Phase 1: WMMA GEMM1 + sigmoid -> lds_p ------------------
    for (int mt = 0; mt < 2; ++mt) {
        const int mr = mt * 16;
        // A regs (contiguous k order: k = 16*h + s, matching B)
        const __bf16* abase = lds_x + (mr + mcol) * X_STRIDE + (hi ? 16 : 0);
        v8bf h0 = *(const v8bf*)(abase);
        v8bf h1 = *(const v8bf*)(abase + 8);
        v8bf h2 = *(const v8bf*)(abase + 32);
        v8bf h3 = *(const v8bf*)(abase + 40);
        v16bf a0, a1;
        #pragma unroll
        for (int i = 0; i < 8; ++i) {
            a0[i] = h0[i]; a0[i + 8] = h1[i];
            a1[i] = h2[i]; a1[i + 8] = h3[i];
        }

        #pragma unroll 2
        for (int j = 0; j < 8; ++j) {
            const int nt   = w * 8 + j;
            const int node = nt * 16 + mcol;
            const __bf16* wp = wb + (size_t)node * D_DIM + (hi ? 16 : 0);
            v16bf b0 = *(const v16bf*)(wp);
            v16bf b1 = *(const v16bf*)(wp + 32);

            v8f acc = {};
            acc = __builtin_amdgcn_wmma_f32_16x16x32_bf16(
                    false, a0, false, b0, (short)0, acc, false, false);
            acc = __builtin_amdgcn_wmma_f32_16x16x32_bf16(
                    false, a1, false, b1, (short)0, acc, false, false);

            const float bb = biasp[node];
            #pragma unroll
            for (int r = 0; r < 8; ++r) {
                float z  = acc[r] + bb;
                float pv = __builtin_amdgcn_rcpf(1.f + __expf(-z));   // fast sigmoid
                int rowM = mr + r + (hi ? 8 : 0);
                lds_p[rowM * P_STRIDE + node] = pv;
            }
        }
    }
    __syncthreads();

    // ------------- Phase 2: path DP, depth-5 subtrees in registers -------------
    // lane = row (0..31); wave w handles subtrees u = 4w .. 4w+3
    {
        const float* prow = lds_p + lane * P_STRIDE;
        __bf16* prout     = lds_path + lane * PATH_STRIDE;

        for (int i = 0; i < 4; ++i) {
            const int u = w * 4 + i;            // depth-5 subtree index 0..31
            // prefix over depths 0..4 (5 loads, mostly broadcast across lanes)
            float pref = 1.f;
            int node = 0;
            #pragma unroll
            for (int d = 0; d < 5; ++d) {
                int bit  = (u >> (4 - d)) & 1;
                float pv = prow[node];
                pref    *= bit ? pv : (1.f - pv);
                node     = 2 * node + 1 + bit;  // ends at 31 + u
            }
            // expand subtree rooted at `node` (depth 5) fully in registers
            float a1[2], a2[4], a3[8], a4[16], a5[32];
            {
                float pv = prow[node];
                a1[0] = pref * (1.f - pv); a1[1] = pref * pv;
            }
            #pragma unroll
            for (int k = 0; k < 2; ++k) {
                float pv = prow[(node << 1) + 1 + k];
                a2[2*k] = a1[k] * (1.f - pv); a2[2*k+1] = a1[k] * pv;
            }
            #pragma unroll
            for (int k = 0; k < 4; ++k) {
                float pv = prow[(node << 2) + 3 + k];
                a3[2*k] = a2[k] * (1.f - pv); a3[2*k+1] = a2[k] * pv;
            }
            #pragma unroll
            for (int k = 0; k < 8; ++k) {
                float pv = prow[(node << 3) + 7 + k];
                a4[2*k] = a3[k] * (1.f - pv); a4[2*k+1] = a3[k] * pv;
            }
            #pragma unroll
            for (int k = 0; k < 16; ++k) {
                float pv = prow[(node << 4) + 15 + k];
                a5[2*k] = a4[k] * (1.f - pv); a5[2*k+1] = a4[k] * pv;
            }
            // write 32 bf16 path values (leaves u*32 .. u*32+31), packed pairs
            #pragma unroll
            for (int k = 0; k < 16; ++k) {
                unsigned pk = (unsigned)f2bfbits(a5[2*k])
                            | ((unsigned)f2bfbits(a5[2*k+1]) << 16);
                *(unsigned*)(prout + u * 32 + 2 * k) = pk;
            }
        }
    }
    __syncthreads();

    // ------------------ Phase 3: out = path @ Q via WMMA ------------------
    // wave w: mtile = w&1, K-chunks (w>>1)*8 .. +7  (4 waves per mtile)
    {
        const int mt  = w & 1;
        const int mr  = mt * 16;
        const int kc0 = (w >> 1) * 8;

        v8f acc = {};
        #pragma unroll 2
        for (int j = 0; j < 8; ++j) {
            const int kc = kc0 + j;
            const __bf16* ab = lds_path + (mr + mcol) * PATH_STRIDE + kc * 32 + (hi ? 16 : 0);
            v8bf h0 = *(const v8bf*)(ab);
            v8bf h1 = *(const v8bf*)(ab + 8);
            const __bf16* bb = lds_qt + mcol * QT_STRIDE + kc * 32 + (hi ? 16 : 0);
            v8bf g0 = *(const v8bf*)(bb);
            v8bf g1 = *(const v8bf*)(bb + 8);
            v16bf av, bv;
            #pragma unroll
            for (int i2 = 0; i2 < 8; ++i2) {
                av[i2] = h0[i2]; av[i2 + 8] = h1[i2];
                bv[i2] = g0[i2]; bv[i2 + 8] = g1[i2];
            }
            acc = __builtin_amdgcn_wmma_f32_16x16x32_bf16(
                    false, av, false, bv, (short)0, acc, false, false);
        }
        #pragma unroll
        for (int r = 0; r < 8; ++r) {
            int rowM = mr + r + (hi ? 8 : 0);
            atomicAdd(&lds_o[rowM * C_DIM + mcol], acc[r]);
        }
    }
    __syncthreads();

    for (int i = t; i < TB * C_DIM; i += 256)
        out[(size_t)row0 * C_DIM + i] = lds_o[i];
}

// ---------------------------------------------------------------------------
extern "C" void kernel_launch(void* const* d_in, const int* in_sizes, int n_in,
                              void* d_out, int out_size, void* d_ws, size_t ws_size,
                              hipStream_t stream) {
    (void)in_sizes; (void)n_in; (void)out_size; (void)ws_size;

    const float* X    = (const float*)d_in[0];
    const float* W    = (const float*)d_in[1];
    const float* b    = (const float*)d_in[2];
    const float* beta = (const float*)d_in[3];
    const float* leaf = (const float*)d_in[4];
    float* out = (float*)d_out;

    unsigned short* qt = (unsigned short*)((char*)d_ws + WS_QT_OFF);
    float* biasp       = (float*)((char*)d_ws + WS_BIAS_OFF);
    __bf16* wbuf       = (__bf16*)((char*)d_ws + WS_WB_OFF);

    sdt_prep_kernel<<<64, 256, 0, stream>>>(W, b, beta, leaf, qt, biasp, wbuf);

    const size_t smem = 237312;   // p + path + Q^T + X + out accumulators
    sdt_main_kernel<<<B_TOTAL / TB, 256, smem, stream>>>(
        X, (const __bf16*)qt, biasp, wbuf, out);
}
